// BlockSparseAttention_17927193493838
// MI455X (gfx1250) — matrix operand order repaired and checked by reading, not compiled, not run on hardware
//
#include <hip/hip_runtime.h>
#include <hip/hip_bf16.h>

typedef __bf16 bf16x8 __attribute__((ext_vector_type(8)));
typedef __bf16 v16bf  __attribute__((ext_vector_type(16)));
typedef float  v8f    __attribute__((ext_vector_type(8)));

#define HIDDEN 2048
#define NHEADS 16
#define HDIM   128
#define SEQ    2048
#define BATCH  2
#define MROWS  (BATCH * SEQ)   // 4096

static __device__ __forceinline__ v16bf cat8(bf16x8 lo, bf16x8 hi) {
  return __builtin_shufflevector(lo, hi, 0, 1, 2, 3, 4, 5, 6, 7, 8, 9, 10, 11, 12, 13, 14, 15);
}

static __device__ __forceinline__ v8f vzero8() {
  v8f z;
#pragma unroll
  for (int i = 0; i < 8; ++i) z[i] = 0.0f;
  return z;
}

static __device__ __forceinline__ v8f wmma_bf16(v16bf a, v16bf b, v8f c) {
  // (neg_a, A, neg_b, B, c_mod, C, reuse_a, reuse_b)
  return __builtin_amdgcn_wmma_f32_16x16x32_bf16(false, a, false, b, (short)0, c, false, false);
}

// ---------------------------------------------------------------- convert X
__global__ void cvt_bf16_k(const float* __restrict__ in, __bf16* __restrict__ out, int n) {
  int i = blockIdx.x * blockDim.x + threadIdx.x;
  int stride = gridDim.x * blockDim.x;
  for (; i < n; i += stride) out[i] = (__bf16)in[i];
}

// --------------------------------------------- convert + transpose weights
// W[k][n] fp32 -> Wt[n][k] bf16  (both 2048x2048)
__global__ void wtrans_k(const float* __restrict__ W, __bf16* __restrict__ Wt) {
  __shared__ float t[16][17];
  const int tx = threadIdx.x, ty = threadIdx.y;
  const int k = blockIdx.y * 16 + ty;
  const int n = blockIdx.x * 16 + tx;
  t[ty][tx] = W[(size_t)k * HIDDEN + n];
  __syncthreads();
  const int n2 = blockIdx.x * 16 + ty;
  const int k2 = blockIdx.y * 16 + tx;
  Wt[(size_t)n2 * HIDDEN + k2] = (__bf16)t[tx][ty];
}

// ------------------------------------------------------------- WMMA GEMM
// C[m,n] = sum_k A[m,k] * Bt[n,k];  A: [MROWS,2048] bf16, Bt: [2048,2048] bf16
// Block tile 128(M) x 256(N); 8 waves as 2(M) x 4(N); wave tile 64x64 = 4x4 WMMA tiles.
// Per k-step: 8 fragment loads (16x b128) feed 16 WMMAs (A reused 4x, B reused 4x).
// mode 0: store bf16 row-major [m][n]
// mode 1: store bf16 V-transposed [b][h][d][s]   (n -> (h,d), m -> (b,s))
// mode 2: store fp32 row-major [m][n]
__global__ __launch_bounds__(256) void gemm_bf16_k(const __bf16* __restrict__ A,
                                                   const __bf16* __restrict__ Bt,
                                                   void* __restrict__ out, int mode) {
  const int lane = threadIdx.x & 31;
  const int ln = lane & 15;
  const bool hiH = lane >= 16;
  const int wv = threadIdx.x >> 5;  // 0..7
  const int wn = wv & 3;            // 4 waves along N
  const int wm = wv >> 2;           // 2 waves along M
  const int Mbase = blockIdx.y * 128 + wm * 64;
  const int Nbase = blockIdx.x * 256 + wn * 64;

  v8f acc[4][4];
#pragma unroll
  for (int mt = 0; mt < 4; ++mt)
#pragma unroll
    for (int nt = 0; nt < 4; ++nt) acc[mt][nt] = vzero8();

  // A fragment lane base: row = Mbase+mt*16+ln ; k-runs {0..7,16..23} (+8 for hi half)
  const __bf16* pA[4];
#pragma unroll
  for (int mt = 0; mt < 4; ++mt)
    pA[mt] = A + (size_t)(Mbase + mt * 16 + ln) * HIDDEN + (hiH ? 8 : 0);
  // B fragment lane base: col n = Nbase+nt*16+ln ; k-runs {0..7,16..23} (+8 for hi half)
  const __bf16* pB[4];
#pragma unroll
  for (int nt = 0; nt < 4; ++nt)
    pB[nt] = Bt + (size_t)(Nbase + nt * 16 + ln) * HIDDEN + (hiH ? 8 : 0);

  for (int k = 0; k < HIDDEN; k += 32) {
    v16bf a[4], b[4];
#pragma unroll
    for (int mt = 0; mt < 4; ++mt)
      a[mt] = cat8(*(const bf16x8*)(pA[mt] + k), *(const bf16x8*)(pA[mt] + k + 16));
#pragma unroll
    for (int nt = 0; nt < 4; ++nt)
      b[nt] = cat8(*(const bf16x8*)(pB[nt] + k), *(const bf16x8*)(pB[nt] + k + 16));
#pragma unroll
    for (int mt = 0; mt < 4; ++mt)
#pragma unroll
      for (int nt = 0; nt < 4; ++nt)
        acc[mt][nt] = wmma_bf16(a[mt], b[nt], acc[mt][nt]);
  }

  if (mode == 0) {
    __bf16* O = (__bf16*)out;
#pragma unroll
    for (int mt = 0; mt < 4; ++mt)
#pragma unroll
      for (int nt = 0; nt < 4; ++nt) {
        const int col = Nbase + nt * 16 + ln;
#pragma unroll
        for (int j = 0; j < 8; ++j) {
          const int row = Mbase + mt * 16 + (hiH ? 8 : 0) + j;
          O[(size_t)row * HIDDEN + col] = (__bf16)acc[mt][nt][j];
        }
      }
  } else if (mode == 1) {
    __bf16* Vt = (__bf16*)out;
#pragma unroll
    for (int mt = 0; mt < 4; ++mt)
#pragma unroll
      for (int nt = 0; nt < 4; ++nt) {
        const int n = Nbase + nt * 16 + ln;
        const int hh = n >> 7, d = n & (HDIM - 1);
        const int srow = Mbase + mt * 16 + (hiH ? 8 : 0);
        const int bb = srow >> 11, sl = srow & (SEQ - 1);
        bf16x8 v;
#pragma unroll
        for (int j = 0; j < 8; ++j) v[j] = (__bf16)acc[mt][nt][j];
        *(bf16x8*)(Vt + ((size_t)(bb * NHEADS + hh) * HDIM + d) * SEQ + sl) = v;
      }
  } else {
    float* O = (float*)out;
#pragma unroll
    for (int mt = 0; mt < 4; ++mt)
#pragma unroll
      for (int nt = 0; nt < 4; ++nt) {
        const int col = Nbase + nt * 16 + ln;
#pragma unroll
        for (int j = 0; j < 8; ++j) {
          const int row = Mbase + mt * 16 + (hiH ? 8 : 0) + j;
          O[(size_t)row * HIDDEN + col] = acc[mt][nt][j];
        }
      }
  }
}

// --------------------------------------------- block-banded flash attention
// grid: (SEQ/64, NHEADS, BATCH), block 128 (4 waves, 16 query rows each)
__global__ __launch_bounds__(128) void attn_k(const __bf16* __restrict__ Q,
                                              const __bf16* __restrict__ K,
                                              const __bf16* __restrict__ Vt,
                                              __bf16* __restrict__ O) {
  __shared__ __bf16 plds[4][16][32];  // per-wave P staging (C-layout -> A-layout)
  const int qb = blockIdx.x, h = blockIdx.y, b = blockIdx.z;
  const int wv = threadIdx.x >> 5;
  const int lane = threadIdx.x & 31, ln = lane & 15;
  const bool hiH = lane >= 16;
  const int mofs = hiH ? 8 : 0;
  const int q0 = qb * 64 + wv * 16;

  // Q fragments over D=128 (4 chunks of k=32)
  const __bf16* qp = Q + ((size_t)(b * SEQ + q0 + ln)) * HIDDEN + h * HDIM + (hiH ? 8 : 0);
  v16bf qf[4];
#pragma unroll
  for (int dc = 0; dc < 4; ++dc)
    qf[dc] = cat8(*(const bf16x8*)(qp + dc * 32), *(const bf16x8*)(qp + dc * 32 + 16));

  float m[8], l[8];
  v8f o[8];
#pragma unroll
  for (int j = 0; j < 8; ++j) { m[j] = -1e30f; l[j] = 0.0f; }
#pragma unroll
  for (int dt = 0; dt < 8; ++dt) o[dt] = vzero8();

  const float scale = 0.0883883476483184f;  // 128^-0.5
  const int kbLo = (qb >= 2) ? (qb - 2) : 0;
  const int kbHi = (qb + 2 <= (SEQ / 64 - 1)) ? (qb + 2) : (SEQ / 64 - 1);

  for (int kb = kbLo; kb <= kbHi; ++kb) {
    for (int nc = 0; nc < 2; ++nc) {
      const int key0 = kb * 64 + nc * 32;
      v8f s0 = vzero8(), s1 = vzero8();
      const __bf16* kp0 = K + ((size_t)(b * SEQ + key0 + ln)) * HIDDEN + h * HDIM + (hiH ? 8 : 0);
      const __bf16* kp1 = kp0 + (size_t)16 * HIDDEN;
#pragma unroll
      for (int dc = 0; dc < 4; ++dc) {
        v16bf kf0 = cat8(*(const bf16x8*)(kp0 + dc * 32), *(const bf16x8*)(kp0 + dc * 32 + 16));
        s0 = wmma_bf16(qf[dc], kf0, s0);
        v16bf kf1 = cat8(*(const bf16x8*)(kp1 + dc * 32), *(const bf16x8*)(kp1 + dc * 32 + 16));
        s1 = wmma_bf16(qf[dc], kf1, s1);
      }
      s0 = s0 * scale;
      s1 = s1 * scale;

      float alpha[8];
#pragma unroll
      for (int j = 0; j < 8; ++j) {
        float rm = fmaxf(s0[j], s1[j]);
        rm = fmaxf(rm, __shfl_xor(rm, 1));
        rm = fmaxf(rm, __shfl_xor(rm, 2));
        rm = fmaxf(rm, __shfl_xor(rm, 4));
        rm = fmaxf(rm, __shfl_xor(rm, 8));   // within 16-lane half -> full row max
        const float nm = fmaxf(m[j], rm);
        alpha[j] = __expf(m[j] - nm);
        m[j] = nm;
        const float p0 = __expf(s0[j] - nm);
        const float p1 = __expf(s1[j] - nm);
        float rs = p0 + p1;
        rs += __shfl_xor(rs, 1);
        rs += __shfl_xor(rs, 2);
        rs += __shfl_xor(rs, 4);
        rs += __shfl_xor(rs, 8);
        l[j] = l[j] * alpha[j] + rs;
        plds[wv][j + mofs][ln] = (__bf16)p0;
        plds[wv][j + mofs][ln + 16] = (__bf16)p1;
      }
#pragma unroll
      for (int dt = 0; dt < 8; ++dt)
#pragma unroll
        for (int j = 0; j < 8; ++j) o[dt][j] *= alpha[j];

      asm volatile("s_wait_dscnt 0" ::: "memory");
      const __bf16* pp = &plds[wv][ln][hiH ? 8 : 0];
      v16bf pf = cat8(*(const bf16x8*)pp, *(const bf16x8*)(pp + 16));

#pragma unroll
      for (int dt = 0; dt < 8; ++dt) {
        const __bf16* vp =
            Vt + ((size_t)((b * NHEADS + h) * HDIM + dt * 16 + ln)) * SEQ + key0 + (hiH ? 8 : 0);
        v16bf vf = cat8(*(const bf16x8*)vp, *(const bf16x8*)(vp + 16));
        o[dt] = wmma_bf16(pf, vf, o[dt]);
      }
    }
  }

#pragma unroll
  for (int j = 0; j < 8; ++j) {
    const float inv = 1.0f / l[j];
    const size_t row = (size_t)(b * SEQ + q0 + j + mofs) * HIDDEN + h * HDIM;
#pragma unroll
    for (int dt = 0; dt < 8; ++dt)
      O[row + dt * 16 + ln] = (__bf16)(o[dt][j] * inv);
  }
}

// ---------------------------------------------------------------- launcher
extern "C" void kernel_launch(void* const* d_in, const int* in_sizes, int n_in,
                              void* d_out, int out_size, void* d_ws, size_t ws_size,
                              hipStream_t stream) {
  (void)in_sizes; (void)n_in; (void)out_size; (void)ws_size;
  const float* X  = (const float*)d_in[0];
  const float* Wq = (const float*)d_in[1];
  const float* Wk = (const float*)d_in[2];
  const float* Wv = (const float*)d_in[3];
  const float* Wo = (const float*)d_in[4];

  __bf16* ws = (__bf16*)d_ws;
  size_t off = 0;
  __bf16* Xbf = ws + off; off += (size_t)MROWS * HIDDEN;
  __bf16* Wqt = ws + off; off += (size_t)HIDDEN * HIDDEN;
  __bf16* Wkt = ws + off; off += (size_t)HIDDEN * HIDDEN;
  __bf16* Wvt = ws + off; off += (size_t)HIDDEN * HIDDEN;
  __bf16* Wot = ws + off; off += (size_t)HIDDEN * HIDDEN;
  __bf16* Qb  = ws + off; off += (size_t)MROWS * HIDDEN;
  __bf16* Kb  = ws + off; off += (size_t)MROWS * HIDDEN;
  __bf16* Vt  = ws + off; off += (size_t)MROWS * HIDDEN;  // [B][H][D][S]
  __bf16* Ob  = ws + off; off += (size_t)MROWS * HIDDEN;

  cvt_bf16_k<<<4096, 256, 0, stream>>>(X, Xbf, MROWS * HIDDEN);

  dim3 tg(HIDDEN / 16, HIDDEN / 16), tb(16, 16);
  wtrans_k<<<tg, tb, 0, stream>>>(Wq, Wqt);
  wtrans_k<<<tg, tb, 0, stream>>>(Wk, Wkt);
  wtrans_k<<<tg, tb, 0, stream>>>(Wv, Wvt);
  wtrans_k<<<tg, tb, 0, stream>>>(Wo, Wot);

  dim3 gg(HIDDEN / 256, MROWS / 128);  // (8, 32)
  gemm_bf16_k<<<gg, 256, 0, stream>>>(Xbf, Wqt, (void*)Qb, 0);
  gemm_bf16_k<<<gg, 256, 0, stream>>>(Xbf, Wkt, (void*)Kb, 0);
  gemm_bf16_k<<<gg, 256, 0, stream>>>(Xbf, Wvt, (void*)Vt, 1);

  dim3 ag(SEQ / 64, NHEADS, BATCH);
  attn_k<<<ag, 128, 0, stream>>>(Qb, Kb, Vt, Ob);

  gemm_bf16_k<<<gg, 256, 0, stream>>>(Ob, Wot, d_out, 2);
}
